// ReLUSqFFN_63307817943750
// MI455X (gfx1250) — 8-bit matrix operand order repaired and checked by reading, compiled, not run on hardware
//
#include <hip/hip_runtime.h>
#include <hip/hip_bf16.h>
#include <stdint.h>


typedef __attribute__((ext_vector_type(8))) int v8i;
typedef __attribute__((ext_vector_type(4))) int v4i;
typedef __attribute__((ext_vector_type(2))) int v2i;

typedef const int8_t __attribute__((address_space(1)))* gcp8;
typedef const v2i    __attribute__((address_space(1)))* gcp2;
typedef const v4i    __attribute__((address_space(1)))* gcp4;
typedef const float  __attribute__((address_space(1)))* gcpf;
typedef float        __attribute__((address_space(1)))* gpf;

#define EPSF 1e-5f

static constexpr size_t NW        = 8192ull * 2048ull;
static constexpr size_t OFF_SC    = 0;
static constexpr size_t OFF_XQ    = 256;
static constexpr size_t OFF_WQ1   = OFF_XQ  + NW;
static constexpr size_t OFF_WQ2   = OFF_WQ1 + NW;
static constexpr size_t OFF_HQ    = OFF_WQ2 + NW;
static constexpr size_t OFF_AINV  = OFF_HQ + 8192ull * 8192ull;
static constexpr size_t OFF_BINV  = OFF_AINV + 8192ull * 4ull;
static constexpr size_t OFF_H     = OFF_BINV + 8192ull * 4ull;


__global__ void k_zero_scalars(float* sc) {
    if (threadIdx.x < 8) sc[threadIdx.x] = 0.0f;
}

__global__ __launch_bounds__(256) void k_absmean(const float* __restrict__ W,
                                                 float* __restrict__ out,
                                                 size_t n) {
    float s = 0.0f;
    for (size_t i = (size_t)blockIdx.x * blockDim.x + threadIdx.x; i < n;
         i += (size_t)gridDim.x * blockDim.x)
        s += fabsf(W[i]);
    __shared__ float sm[256];
    sm[threadIdx.x] = s;
    __syncthreads();
    for (int off = 128; off > 0; off >>= 1) {
        if ((int)threadIdx.x < off) sm[threadIdx.x] += sm[threadIdx.x + off];
        __syncthreads();
    }
    if (threadIdx.x == 0) atomicAdd(out, sm[0]);
}

__global__ void k_finalize(float* sc) {
    if (threadIdx.x == 0) {
        float m1 = sc[0] / (float)NW;
        float d1 = fmaxf(m1, EPSF);
        sc[2] = 1.0f / d1;
        sc[3] = d1;
        float m2 = sc[1] / (float)NW;
        float d2 = fmaxf(m2, EPSF);
        sc[4] = 1.0f / d2;
        sc[5] = d2;
    }
}

__global__ __launch_bounds__(256) void k_wquant(const float* __restrict__ W,
                                                int8_t* __restrict__ Q,
                                                const float* __restrict__ s_ptr,
                                                size_t n) {
    float s = *s_ptr;
    for (size_t i = (size_t)blockIdx.x * blockDim.x + threadIdx.x; i < n;
         i += (size_t)gridDim.x * blockDim.x) {
        float v = rintf(W[i] * s);
        v = fminf(fmaxf(v, -1.0f), 1.0f);
        Q[i] = (int8_t)v;
    }
}

__global__ __launch_bounds__(256) void k_rmsnorm_quant(const float* __restrict__ X,
                                                       int8_t* __restrict__ Q,
                                                       float* __restrict__ rinv,
                                                       int L) {
    const int row = blockIdx.x;
    const float* xr = X + (size_t)row * L;
    float ss = 0.0f, mx = 0.0f;
    for (int i = threadIdx.x; i < L; i += 256) {
        float v = xr[i];
        ss += v * v;
        mx = fmaxf(mx, fabsf(v));
    }
    __shared__ float s_ss[256];
    __shared__ float s_mx[256];
    s_ss[threadIdx.x] = ss;
    s_mx[threadIdx.x] = mx;
    __syncthreads();
    for (int off = 128; off > 0; off >>= 1) {
        if ((int)threadIdx.x < off) {
            s_ss[threadIdx.x] += s_ss[threadIdx.x + off];
            s_mx[threadIdx.x] = fmaxf(s_mx[threadIdx.x], s_mx[threadIdx.x + off]);
        }
        __syncthreads();
    }
    const float rs   = rsqrtf(s_ss[0] / (float)L + EPSF);
    const float amax = fmaxf(s_mx[0] * rs, EPSF);
    const float scale = 127.0f / amax;
    if (threadIdx.x == 0) rinv[row] = amax / 127.0f;
    int8_t* qr = Q + (size_t)row * L;
    for (int i = threadIdx.x; i < L; i += 256) {
        float v = rintf(xr[i] * rs * scale);
        v = fminf(fmaxf(v, -128.0f), 127.0f);
        qr[i] = (int8_t)v;
    }
}

__global__ __launch_bounds__(256) void k_gemm_iu8(const int8_t* __restrict__ A_,
                                                  const int8_t* __restrict__ B_,
                                                  const float* __restrict__ row_inv_,
                                                  const float* __restrict__ w_deq,
                                                  float* __restrict__ C_,
                                                  int M, int N, int K, int relu2) {
    const int lane  = threadIdx.x & 31;
    const int wave  = threadIdx.x >> 5;
    const int wm    = wave & 3;
    const int wn    = wave >> 2;
    const int lrow  = lane & 15;
    const int lhalf = lane >> 4;

    const int m_base = blockIdx.y * 128 + wm * 32;
    const int n_base = blockIdx.x * 128 + wn * 64;

    gcp8 A = (gcp8)A_;
    gcp8 B = (gcp8)B_;
    gcpf row_inv = (gcpf)row_inv_;
    gpf  C = (gpf)C_;

    gcp8 apt[2];
    gcp8 bpt[4];
#pragma unroll
    for (int t = 0; t < 2; ++t)
        apt[t] = A + (size_t)(m_base + t * 16 + lrow) * K + lhalf * 16;
#pragma unroll
    for (int u = 0; u < 4; ++u)
        bpt[u] = B + (size_t)(n_base + u * 16 + lrow) * K + lhalf * 16;

    auto loadA = [](gcp8 ap) -> v8i {
        v4i d0 = *(gcp4)(ap + 0);
        v4i d1 = *(gcp4)(ap + 32);
        return (v8i){d0.x, d0.y, d0.z, d0.w, d1.x, d1.y, d1.z, d1.w};
    };
    auto loadB = [](gcp8 bp) -> v8i {
        v4i e0 = *(gcp4)(bp + 0);
        v4i e1 = *(gcp4)(bp + 32);
        return (v8i){e0.x, e0.y, e0.z, e0.w, e1.x, e1.y, e1.z, e1.w};
    };

    v8i acc[2][4] = {};
    v8i a0[2], b0[4], a1[2], b1[4];

#pragma unroll
    for (int t = 0; t < 2; ++t) a0[t] = loadA(apt[t]);
#pragma unroll
    for (int u = 0; u < 4; ++u) b0[u] = loadB(bpt[u]);

    const int steps = K >> 7;
    for (int it = 0; it < steps; ++it) {
#pragma unroll
        for (int t = 0; t < 2; ++t) a1[t] = loadA(apt[t] + 64);
#pragma unroll
        for (int u = 0; u < 4; ++u) b1[u] = loadB(bpt[u] + 64);
        __builtin_prefetch((const void*)(apt[0] + 256));
        __builtin_prefetch((const void*)(apt[1] + 256));

#pragma unroll
        for (int t = 0; t < 2; ++t)
#pragma unroll
            for (int u = 0; u < 4; ++u)
                acc[t][u] = __builtin_amdgcn_wmma_i32_16x16x64_iu8(
                     true, a0[t],  true, b0[u], acc[t][u],
                     false,  false);

#pragma unroll
        for (int t = 0; t < 2; ++t) apt[t] += 128;
#pragma unroll
        for (int u = 0; u < 4; ++u) bpt[u] += 128;
#pragma unroll
        for (int t = 0; t < 2; ++t) a0[t] = loadA(apt[t]);
#pragma unroll
        for (int u = 0; u < 4; ++u) b0[u] = loadB(bpt[u]);

#pragma unroll
        for (int t = 0; t < 2; ++t)
#pragma unroll
            for (int u = 0; u < 4; ++u)
                acc[t][u] = __builtin_amdgcn_wmma_i32_16x16x64_iu8(
                     true, a1[t],  true, b1[u], acc[t][u],
                     false,  false);
    }

    const float wd = *w_deq;
#pragma unroll
    for (int t = 0; t < 2; ++t) {
        const int gm0 = m_base + t * 16 + lhalf * 8;
#pragma unroll
        for (int u = 0; u < 4; ++u) {
            const int gn = n_base + u * 16 + lrow;
#pragma unroll
            for (int r = 0; r < 8; ++r) {
                const int gm = gm0 + r;
                float f = (float)acc[t][u][r] * row_inv[gm] * wd;
                if (relu2) { f = fmaxf(f, 0.0f); f = f * f; }
                C[(size_t)gm * N + gn] = f;
            }
        }
    }
}


extern "C" void kernel_launch(void* const* d_in, const int* in_sizes, int n_in,
                              void* d_out, int out_size, void* d_ws, size_t ws_size,
                              hipStream_t stream) {
    const float* x  = (const float*)d_in[0];
    const float* W1 = (const float*)d_in[1];
    const float* W2 = (const float*)d_in[2];
    float* out = (float*)d_out;

    char* ws = (char*)d_ws;
    float*  sc   = (float*)(ws + OFF_SC);
    int8_t* Xq   = (int8_t*)(ws + OFF_XQ);
    int8_t* Wq1  = (int8_t*)(ws + OFF_WQ1);
    int8_t* Wq2  = (int8_t*)(ws + OFF_WQ2);
    int8_t* Hq   = (int8_t*)(ws + OFF_HQ);
    float*  ainv = (float*)(ws + OFF_AINV);
    float*  binv = (float*)(ws + OFF_BINV);
    float*  H    = (float*)(ws + OFF_H);

    const int M = 8192;
    const int D = 2048;
    const int F = 8192;

    k_zero_scalars<<<1, 32, 0, stream>>>(sc);
    k_absmean<<<2048, 256, 0, stream>>>(W1, sc + 0, NW);
    k_absmean<<<2048, 256, 0, stream>>>(W2, sc + 1, NW);
    k_finalize<<<1, 32, 0, stream>>>(sc);

    k_wquant<<<8192, 256, 0, stream>>>(W1, Wq1, sc + 2, NW);
    k_wquant<<<8192, 256, 0, stream>>>(W2, Wq2, sc + 4, NW);

    k_rmsnorm_quant<<<M, 256, 0, stream>>>(x, Xq, ainv, D);

    k_gemm_iu8<<<dim3(F / 128, M / 128), 256, 0, stream>>>(
        Xq, Wq1, ainv, sc + 3, H, M, F, D,  1);

    k_rmsnorm_quant<<<M, 256, 0, stream>>>(H, Hq, binv, F);

    k_gemm_iu8<<<dim3(D / 128, M / 128), 256, 0, stream>>>(
        Hq, Wq2, binv, sc + 5, out, M, D, F,  0);
}
